// Generator_17437567222067
// MI455X (gfx1250) — matrix operand order repaired and checked by reading, not compiled, not run on hardware
//
#include <hip/hip_runtime.h>
#include <hip/hip_bf16.h>
#include <cstdint>
#include <cstddef>

// Problem constants
#define HDIM  300      // hidden size H
#define HPAD  320      // K padded to multiple of 32 (WMMA K=32 chunks)
#define G3    900      // 3*H
#define HGSTR 912      // hg LDS row stride (57 tiles * 16)
#define BB    64       // batch
#define TU_   50
#define TT_   64
#define VV    19495    // vocab

typedef __attribute__((ext_vector_type(16))) __bf16       v16bf;
typedef __attribute__((ext_vector_type(8)))  __bf16       v8bf;
typedef __attribute__((ext_vector_type(8)))  float        v8f;
typedef __attribute__((ext_vector_type(4)))  unsigned int u32x4;

// ---------- small helpers ----------
__device__ __forceinline__ float sigmoidf_(float x) { return 1.f / (1.f + expf(-x)); }

// pack 8 f32 -> 8 bf16 and store as one 16B chunk (LDS b128 store)
__device__ __forceinline__ void store_bf8(__bf16* dst, float4 a, float4 b) {
  v8bf t;
  t[0] = (__bf16)a.x; t[1] = (__bf16)a.y; t[2] = (__bf16)a.z; t[3] = (__bf16)a.w;
  t[4] = (__bf16)b.x; t[5] = (__bf16)b.y; t[6] = (__bf16)b.z; t[7] = (__bf16)b.w;
  *(v8bf*)dst = t;
}

// Load a v16bf WMMA fragment as two 16-byte chunks (layouts per CDNA5 ISA):
//  A-frag (16x32, M=lane&15): 8 contiguous bf16 at k=hi*8, then 8 more at k=16+hi*8 -> delta=16
//  B-frag (32x16, N=lane&15): 8 contiguous bf16 at k=hi*8, then 8 more at k=16+hi*8 -> delta=16
__device__ __forceinline__ v16bf load_frag(const __bf16* p, int deltaElems) {
  union { v16bf v; u32x4 q[2]; } u;
  u.q[0] = *(const u32x4*)(p);
  u.q[1] = *(const u32x4*)(p + deltaElems);
  return u.v;
}
__device__ __forceinline__ v8f wmma_bf16(v16bf a, v16bf b, v8f c) {
  return __builtin_amdgcn_wmma_f32_16x16x32_bf16(false, a, false, b, (short)0, c, false, false);
}

// CDNA5 async global->LDS copy (ASYNCcnt-tracked), CDNA5 ISA
__device__ __forceinline__ void async_b128(void* lds, const void* gaddr) {
  asm volatile("global_load_async_to_lds_b128 %0, %1, off"
               :: "v"((unsigned)(uintptr_t)lds),
                  "v"((unsigned long long)(uintptr_t)gaddr)
               : "memory");
}
__device__ __forceinline__ void wait_async0() {
  asm volatile("s_wait_asynccnt 0x0" ::: "memory");
}

// ---------- 1) f32 -> bf16 weight conversion with K padding ----------
__global__ void convert_pad_kernel(const float* __restrict__ src, __bf16* __restrict__ dst,
                                   int R, int K0, int Kp) {
  int n = R * Kp;
  for (int j = blockIdx.x * blockDim.x + threadIdx.x; j < n; j += gridDim.x * blockDim.x) {
    int r = j / Kp, k = j - r * Kp;
    dst[j] = (k < K0) ? (__bf16)src[r * K0 + k] : (__bf16)0.f;
  }
}

// ---------- 2) x-gates GEMM: out[M,900] = gather/emb(A)[M,300] @ Wih^T + bih ----------
// Fuses embedding gather into the A-tile staging. bf16 WMMA, f32 accumulate.
__global__ __launch_bounds__(256)
void xgates_kernel(const float* __restrict__ Asrc, const int* __restrict__ tokens,
                   const float* __restrict__ emb,
                   const float* __restrict__ W,    // [900][300] row-major (N-major, K contiguous)
                   const float* __restrict__ bias, // [900]
                   float* __restrict__ out)        // [M][900]
{
  __shared__ __attribute__((aligned(16))) __bf16 As[128 * 32];
  __shared__ __attribute__((aligned(16))) __bf16 Bs[64 * 32];
  const int tid = threadIdx.x, lane = tid & 31, wave = tid >> 5;
  const int lo = lane & 15, hi = lane >> 4;
  const int m0 = blockIdx.x * 128, n0 = blockIdx.y * 64;
  const int wm = wave & 3, wn = wave >> 2;   // 4 (M) x 2 (N) waves

  // ---- hoisted per-thread staging pointers (token gather resolved once) ----
  const int acg = tid & 3;                   // A col-group (same for both slots)
  const int ar0 = tid >> 2;                  // A row slot 0 (0..63), slot 1 = +64
  const float* arow[2];
#pragma unroll
  for (int i = 0; i < 2; ++i) {
    int mrow = m0 + ar0 + 64 * i;
    arow[i] = tokens ? (emb + (size_t)tokens[mrow] * HDIM)
                     : (Asrc + (size_t)mrow * HDIM);
  }
  const int br = tid >> 2, bcg = tid & 3;    // B: row (0..63) x col-group
  const int bn = n0 + br;
  const float* brow = W + (size_t)(bn < G3 ? bn : G3 - 1) * HDIM;  // clamped; cols>=900 discarded

  v8f acc[2][2] = {};
  for (int kc = 0; kc < 10; ++kc) {
    const int k0 = kc * 32;
    if (kc < 9) {              // guard-free vector staging (k <= 287 < 300 only for kc<9)
#pragma unroll
      for (int i = 0; i < 2; ++i) {
        const float* p = arow[i] + k0 + acg * 8;
        store_bf8(&As[(ar0 + 64 * i) * 32 + acg * 8],
                  *(const float4*)p, *(const float4*)(p + 4));
      }
      const float* pb = brow + k0 + bcg * 8;
      store_bf8(&Bs[br * 32 + bcg * 8], *(const float4*)pb, *(const float4*)(pb + 4));
    } else {                   // tail chunk: k in [288,320), zero k>=300 branch-free
#pragma unroll
      for (int i = 0; i < 2; ++i) {
        v8bf t;
#pragma unroll
        for (int e = 0; e < 8; ++e) {
          int k = k0 + acg * 8 + e;
          float v = arow[i][k < HDIM ? k : HDIM - 1];
          t[e] = (k < HDIM) ? (__bf16)v : (__bf16)0.f;
        }
        *(v8bf*)&As[(ar0 + 64 * i) * 32 + acg * 8] = t;
      }
      v8bf t;
#pragma unroll
      for (int e = 0; e < 8; ++e) {
        int k = k0 + bcg * 8 + e;
        float v = brow[k < HDIM ? k : HDIM - 1];
        t[e] = (k < HDIM) ? (__bf16)v : (__bf16)0.f;
      }
      *(v8bf*)&Bs[br * 32 + bcg * 8] = t;
    }
    __syncthreads();
    v16bf Af[2], Bf[2];
#pragma unroll
    for (int mt = 0; mt < 2; ++mt)
      Af[mt] = load_frag(&As[(wm * 32 + mt * 16 + lo) * 32 + hi * 8], 16);
#pragma unroll
    for (int nt = 0; nt < 2; ++nt)
      Bf[nt] = load_frag(&Bs[(wn * 32 + nt * 16 + lo) * 32 + hi * 8], 16);
#pragma unroll
    for (int mt = 0; mt < 2; ++mt)
#pragma unroll
      for (int nt = 0; nt < 2; ++nt)
        acc[mt][nt] = wmma_bf16(Af[mt], Bf[nt], acc[mt][nt]);
    __syncthreads();
  }
#pragma unroll
  for (int nt = 0; nt < 2; ++nt) {
    int n = n0 + wn * 32 + nt * 16 + lo;
    if (n < G3) {
      float bb = bias[n];
#pragma unroll
      for (int mt = 0; mt < 2; ++mt)
#pragma unroll
        for (int r = 0; r < 8; ++r) {
          int m = m0 + wm * 32 + mt * 16 + r + hi * 8;
          out[(size_t)m * G3 + n] = acc[mt][nt][r] + bb;
        }
    }
  }
}

// ---------- 3) persistent GRU recurrence (one workgroup runs all T steps) ----------
// 512 threads = 16 waves. Wave w owns N-tiles w*4..w*4+3 (57 valid of 64), all 4 M-tiles.
// h state: f32 in registers; bf16 mirror in LDS (A operand); hg staged bf16 in LDS.
// Whh bf16 fragments are streamed from global each step (576 KB, L2-resident).
__global__ __launch_bounds__(512, 1)
void gru_seq_kernel(const float* __restrict__ xg0, const float* __restrict__ xg1,
                    const int* __restrict__ lens0, const int* __restrict__ lens1,
                    const __bf16* __restrict__ Whh_bf,  // [900][HPAD] bf16, K padded w/ zeros
                    const float* __restrict__ bhh,      // [900]
                    const float* __restrict__ h_init,   // nullptr -> zeros, else [64][300]
                    float* __restrict__ hout0, float* __restrict__ hout1, int hout_stride,
                    __bf16* __restrict__ outs_bf,       // nullptr or [64*T][HPAD] bf16
                    int T)
{
  extern __shared__ __attribute__((aligned(16))) __bf16 smem[];
  __bf16* hbf = smem;                 // [64][HPAD]
  __bf16* hg  = smem + BB * HPAD;     // [64][HGSTR]

  const float* xg   = (blockIdx.x == 0) ? xg0 : xg1;
  const int*   lens = (blockIdx.x == 0) ? lens0 : lens1;
  float*       hout = (blockIdx.x == 0) ? hout0 : hout1;

  const int tid = threadIdx.x, lane = tid & 31, wave = tid >> 5;
  const int lo = lane & 15, hi = lane >> 4;

  // register-resident h state: element j = tid + 512*i
  float hreg[38];
#pragma unroll
  for (int i = 0; i < 38; ++i) {
    int j = tid + 512 * i;
    hreg[i] = (h_init && j < BB * HDIM) ? h_init[j] : 0.f;
  }
  for (int j = tid; j < BB * HPAD; j += 512) hbf[j] = (__bf16)0.f; // zero K-pad once

  // per-wave N-tile metadata (hoisted out of the time loop)
  bool  ntok[4];
  float bbr[4];
  const __bf16* bptr[4];
#pragma unroll
  for (int nt = 0; nt < 4; ++nt) {
    int tIdx = wave * 4 + nt;
    int n = tIdx * 16 + lo;
    int nc = (tIdx < 57) ? (n < G3 ? n : G3 - 1) : 0;
    ntok[nt] = (tIdx < 57);
    bbr[nt]  = (tIdx < 57 && n < G3) ? bhh[n] : 0.f;
    bptr[nt] = Whh_bf + (size_t)nc * HPAD + hi * 8;
  }
  __syncthreads();

  for (int t = 0; t < T; ++t) {
    // publish h -> bf16 LDS (A operand for this step)
#pragma unroll
    for (int i = 0; i < 38; ++i) {
      int j = tid + 512 * i;
      if (j < BB * HDIM) {
        int b = j / HDIM, c = j - b * HDIM;
        hbf[b * HPAD + c] = (__bf16)hreg[i];
      }
    }
    __syncthreads();

    // hg = h @ Whh^T  (bf16 WMMA, f32 accumulate)
    v8f acc[4][4] = {};
    for (int kc = 0; kc < 10; ++kc) {
      const int k0 = kc * 32;
      v16bf Af[4];
#pragma unroll
      for (int mt = 0; mt < 4; ++mt)
        Af[mt] = load_frag(hbf + (mt * 16 + lo) * HPAD + k0 + hi * 8, 16);
#pragma unroll
      for (int nt = 0; nt < 4; ++nt) {
        if (ntok[nt]) {
          v16bf Bf = load_frag(bptr[nt] + k0, 16);
#pragma unroll
          for (int mt = 0; mt < 4; ++mt)
            acc[nt][mt] = wmma_bf16(Af[mt], Bf, acc[nt][mt]);
        }
      }
    }
#pragma unroll
    for (int nt = 0; nt < 4; ++nt) {
      if (ntok[nt]) {
        int n = (wave * 4 + nt) * 16 + lo;
#pragma unroll
        for (int mt = 0; mt < 4; ++mt)
#pragma unroll
          for (int r = 0; r < 8; ++r) {
            int mm = mt * 16 + r + hi * 8;
            hg[mm * HGSTR + n] = (__bf16)(acc[nt][mt][r] + bbr[nt]);
          }
      }
    }
    __syncthreads();

    // gates + state update (f32)
#pragma unroll
    for (int i = 0; i < 38; ++i) {
      int j = tid + 512 * i;
      if (j < BB * HDIM) {
        int b = j / HDIM, c = j - b * HDIM;
        const float* x = xg + (size_t)(b * T + t) * G3;   // xg already includes bih
        float xr = x[c], xz = x[c + HDIM], xn = x[c + 2 * HDIM];
        float hr = (float)hg[b * HGSTR + c];
        float hz = (float)hg[b * HGSTR + c + HDIM];
        float hn = (float)hg[b * HGSTR + c + 2 * HDIM];
        float rg = sigmoidf_(xr + hr);
        float zg = sigmoidf_(xz + hz);
        float ng = tanhf(xn + rg * hn);
        float hnew = (1.f - zg) * ng + zg * hreg[i];
        if (lens) { if (t < lens[b]) hreg[i] = hnew; }
        else      { hreg[i] = hnew; }
        if (outs_bf) outs_bf[(size_t)(b * T + t) * HPAD + c] = (__bf16)hnew;
      }
    }
    if (outs_bf) {  // zero the K-pad of this step's output rows
      for (int j = tid; j < BB * (HPAD - HDIM); j += 512) {
        int b = j / (HPAD - HDIM), c = HDIM + (j - b * (HPAD - HDIM));
        outs_bf[(size_t)(b * T + t) * HPAD + c] = (__bf16)0.f;
      }
    }
    // no barrier needed here: next publish only writes hbf, which was last read
    // before the post-matmul barrier (LDS reads are waitcnt'ed before s_barrier).
    __syncthreads();  // kept between update(read hg) and next matmul(write hg) via publish path
  }

  // final hidden state
#pragma unroll
  for (int i = 0; i < 38; ++i) {
    int j = tid + 512 * i;
    if (j < BB * HDIM) {
      int b = j / HDIM, c = j - b * HDIM;
      hout[(size_t)b * hout_stride + c] = hreg[i];
    }
  }
}

// ---------- 4) vocab projection: logits[4096,19495] = dec_outs @ out_W^T + out_b ----------
// 128x128 macro-tiles, 8 waves as 2(M)x4(N). Operands are pre-converted bf16 (K padded
// to 320 -> no K guards), so staging is a pure copy: double-buffered
// global_load_async_to_lds_b128, overlapped with the WMMAs of the current chunk.
__global__ __launch_bounds__(256)
void proj_kernel(const __bf16* __restrict__ A,   // [4096][HPAD] bf16
                 const __bf16* __restrict__ W,   // [VV][HPAD] bf16
                 const float* __restrict__ bias, // [VV]
                 float* __restrict__ out)        // [4096][VV]
{
  __shared__ __attribute__((aligned(16))) __bf16 As[2][128 * 32];
  __shared__ __attribute__((aligned(16))) __bf16 Bs[2][128 * 32];
  const int tid = threadIdx.x, lane = tid & 31, wave = tid >> 5;
  const int lo = lane & 15, hi = lane >> 4;
  const int m0 = blockIdx.x * 128, n0 = blockIdx.y * 128;
  const int wm = wave >> 2, wn = wave & 3;   // 2 (M) x 4 (N)

  // per-thread staging addresses: rows r0 and r0+64, col-group cg (8 bf16 = 16 B)
  const int r0 = tid >> 2, cg = tid & 3;
  const __bf16* aptr0 = A + (size_t)(m0 + r0) * HPAD + cg * 8;
  const __bf16* aptr1 = A + (size_t)(m0 + r0 + 64) * HPAD + cg * 8;
  int nr0 = n0 + r0;      if (nr0 > VV - 1) nr0 = VV - 1;   // clamped; cols>=VV discarded
  int nr1 = n0 + r0 + 64; if (nr1 > VV - 1) nr1 = VV - 1;
  const __bf16* bptr0 = W + (size_t)nr0 * HPAD + cg * 8;
  const __bf16* bptr1 = W + (size_t)nr1 * HPAD + cg * 8;

  auto stage = [&](int buf, int kc) {
    const int k0 = kc * 32;
    async_b128(&As[buf][r0 * 32 + cg * 8],        aptr0 + k0);
    async_b128(&As[buf][(r0 + 64) * 32 + cg * 8], aptr1 + k0);
    async_b128(&Bs[buf][r0 * 32 + cg * 8],        bptr0 + k0);
    async_b128(&Bs[buf][(r0 + 64) * 32 + cg * 8], bptr1 + k0);
  };

  stage(0, 0);
  wait_async0();
  __syncthreads();

  v8f acc[4][2] = {};
  for (int kc = 0; kc < 10; ++kc) {
    const int cur = kc & 1;
    if (kc < 9) stage(cur ^ 1, kc + 1);          // prefetch next chunk (overlaps WMMAs)
    v16bf Af[4], Bf[2];
#pragma unroll
    for (int mt = 0; mt < 4; ++mt)
      Af[mt] = load_frag(&As[cur][(wm * 64 + mt * 16 + lo) * 32 + hi * 8], 16);
#pragma unroll
    for (int nt = 0; nt < 2; ++nt)
      Bf[nt] = load_frag(&Bs[cur][(wn * 32 + nt * 16 + lo) * 32 + hi * 8], 16);
#pragma unroll
    for (int mt = 0; mt < 4; ++mt)
#pragma unroll
      for (int nt = 0; nt < 2; ++nt)
        acc[mt][nt] = wmma_bf16(Af[mt], Bf[nt], acc[mt][nt]);
    if (kc < 9) { wait_async0(); __syncthreads(); }
  }
#pragma unroll
  for (int nt = 0; nt < 2; ++nt) {
    int n = n0 + wn * 32 + nt * 16 + lo;
    if (n < VV) {
      float bb = bias[n];
#pragma unroll
      for (int mt = 0; mt < 4; ++mt)
#pragma unroll
        for (int r = 0; r < 8; ++r) {
          int m = m0 + wm * 64 + mt * 16 + r + hi * 8;
          __builtin_nontemporal_store(acc[mt][nt][r] + bb, &out[(size_t)m * VV + n]);
        }
    }
  }
}

// ---------- host ----------
extern "C" void kernel_launch(void* const* d_in, const int* in_sizes, int n_in,
                              void* d_out, int out_size, void* d_ws, size_t ws_size,
                              hipStream_t stream) {
  (void)in_sizes; (void)n_in; (void)out_size; (void)ws_size;
  const float* emb      = (const float*)d_in[0];
  const float* enc_Wih  = (const float*)d_in[1];
  const float* enc_Whh  = (const float*)d_in[2];
  const float* enc_bih  = (const float*)d_in[3];
  const float* enc_bhh  = (const float*)d_in[4];
  const float* sess_Wih = (const float*)d_in[5];
  const float* sess_Whh = (const float*)d_in[6];
  const float* sess_bih = (const float*)d_in[7];
  const float* sess_bhh = (const float*)d_in[8];
  const float* dec_Wih  = (const float*)d_in[9];
  const float* dec_Whh  = (const float*)d_in[10];
  const float* dec_bih  = (const float*)d_in[11];
  const float* dec_bhh  = (const float*)d_in[12];
  const float* out_W    = (const float*)d_in[13];
  const float* out_b    = (const float*)d_in[14];
  const int*   u1_in    = (const int*)d_in[15];
  const int*   u1_len   = (const int*)d_in[16];
  const int*   u2_in    = (const int*)d_in[17];
  const int*   u2_len   = (const int*)d_in[18];
  const int*   tgt_in   = (const int*)d_in[19];
  float* logits = (float*)d_out;

  uintptr_t cur = (uintptr_t)d_ws;
  auto alloc = [&](size_t bytes) -> uintptr_t {
    uintptr_t p = cur; cur += (bytes + 255) & ~(size_t)255; return p;
  };
  __bf16* WhhE   = (__bf16*)alloc((size_t)G3 * HPAD * 2);
  __bf16* WhhS   = (__bf16*)alloc((size_t)G3 * HPAD * 2);
  __bf16* WhhD   = (__bf16*)alloc((size_t)G3 * HPAD * 2);
  __bf16* outWbf = (__bf16*)alloc((size_t)VV * HPAD * 2);
  float*  xg_u1  = (float*)alloc((size_t)BB * TU_ * G3 * 4);
  float*  xg_u2  = (float*)alloc((size_t)BB * TU_ * G3 * 4);
  float*  xg_tgt = (float*)alloc((size_t)BB * TT_ * G3 * 4);
  float*  xg_ses = (float*)alloc((size_t)BB * 2 * G3 * 4);
  float*  si     = (float*)alloc((size_t)BB * 2 * HDIM * 4);
  float*  sess_h = (float*)alloc((size_t)BB * HDIM * 4);
  float*  dec_hT = (float*)alloc((size_t)BB * HDIM * 4);
  __bf16* dec_bf = (__bf16*)alloc((size_t)BB * TT_ * HPAD * 2);

  // 1) bf16 weight conversion (K-padded)
  auto cvt = [&](const float* s, __bf16* d, int R) {
    int n = R * HPAD;
    int g = (n + 255) / 256; if (g > 2048) g = 2048;
    convert_pad_kernel<<<g, 256, 0, stream>>>(s, d, R, HDIM, HPAD);
  };
  cvt(enc_Whh, WhhE, G3);
  cvt(sess_Whh, WhhS, G3);
  cvt(dec_Whh, WhhD, G3);
  cvt(out_W, outWbf, VV);

  // 2) x-gate GEMMs with fused embedding gather
  xgates_kernel<<<dim3((BB * TU_) / 128, 15), 256, 0, stream>>>(
      nullptr, u1_in, emb, enc_Wih, enc_bih, xg_u1);
  xgates_kernel<<<dim3((BB * TU_) / 128, 15), 256, 0, stream>>>(
      nullptr, u2_in, emb, enc_Wih, enc_bih, xg_u2);
  xgates_kernel<<<dim3((BB * TT_) / 128, 15), 256, 0, stream>>>(
      nullptr, tgt_in, emb, dec_Wih, dec_bih, xg_tgt);

  // 3) recurrences (persistent workgroups; ~154 KB dynamic LDS)
  size_t shmem = (size_t)(BB * HPAD + BB * HGSTR) * sizeof(__bf16);
  hipFuncSetAttribute((const void*)gru_seq_kernel,
                      hipFuncAttributeMaxDynamicSharedMemorySize, (int)shmem);
  // both utterance encoders concurrently (grid=2); h -> si[b][slot][:]
  gru_seq_kernel<<<2, 512, shmem, stream>>>(xg_u1, xg_u2, u1_len, u2_len, WhhE, enc_bhh,
                                            nullptr, si, si + HDIM, 2 * HDIM, nullptr, TU_);
  // session GRU over si (rows m = b*2+t)
  xgates_kernel<<<dim3(1, 15), 256, 0, stream>>>(si, nullptr, nullptr, sess_Wih, sess_bih, xg_ses);
  gru_seq_kernel<<<1, 512, shmem, stream>>>(xg_ses, xg_ses, nullptr, nullptr, WhhS, sess_bhh,
                                            nullptr, sess_h, sess_h, HDIM, nullptr, 2);
  // teacher-forced decoder; emits bf16 outputs for the projection
  gru_seq_kernel<<<1, 512, shmem, stream>>>(xg_tgt, xg_tgt, nullptr, nullptr, WhhD, dec_bhh,
                                            sess_h, dec_hT, dec_hT, HDIM, dec_bf, TT_);

  // 4) vocab projection -> logits [B,TT,V]
  proj_kernel<<<dim3((BB * TT_) / 128, (VV + 127) / 128), 256, 0, stream>>>(
      dec_bf, outWbf, out_b, logits);
}
